// DocumentAttnOverAttn_77223511982527
// MI455X (gfx1250) — matrix operand order repaired and checked by reading, not compiled, not run on hardware
//
#include <hip/hip_runtime.h>

// ---------------------------------------------------------------------------
// Fused attention-over-attention + gated fusion for MI455X (gfx1250, wave32).
// Flash-style single pass (no [B,L,L] materialization). All three GEMMs
// (QK^T, P@V, z@W_f) run on v_wmma_f32_16x16x32_bf16. LDS keeps both a
// row-major and a transposed (padded, 16B-aligned rows) copy of each staged
// chunk so every WMMA fragment load is a ds_load_b128. Next-chunk
// global_prefetch_b8 overlaps staging latency with WMMA work.
// ---------------------------------------------------------------------------

typedef __attribute__((ext_vector_type(16))) __bf16 v16bf;
typedef __attribute__((ext_vector_type(8)))  __bf16 v8bf;
typedef __attribute__((ext_vector_type(4)))  __bf16 v4bf;
typedef __attribute__((ext_vector_type(8)))  float  v8f;

#define Bn    8
#define Ld    2048
#define Dd    256
#define ROWS  32      // query rows per workgroup
#define TPB   64      // 2 waves (wave32)
#define NCH   32      // key-chunk size
#define KF    1024    // fusion K = 4*D
#define TPAD  40      // transposed-row stride in halfs (80 B = 5 x 16 B)

// ---- shared-memory pool layout (bytes) ----
// [0      , 20480) : T   : transposed chunk  [256][TPAD] bf16  (c1^T, then Wf^T)
// [20480  , 36864) : ROW : row-major chunk   [32][256]  bf16   (c1 chunk) -> aug_c2 in epilogue
// [36864  , 53248) : Q   : query tile        [32][256]  bf16
// [53248  , 55296) : P   : softmax scratch   [2][16][32] bf16
// [55296  , 55552) : GP  : gate partials     [2][32] f32
// [55552  , 55680) : G   : gate              [32] f32
#define OFF_T    0
#define OFF_ROW  20480
#define OFF_Q    36864
#define OFF_P    53248
#define OFF_GP   55296
#define OFF_G    55552
#define POOLSZ   55680

__global__ __launch_bounds__(TPB)
void aoa_fused_kernel(const float* __restrict__ c1,
                      const float* __restrict__ c2,
                      const unsigned char* __restrict__ cmask,
                      const float* __restrict__ Wf,
                      const float* __restrict__ bfv,
                      const float* __restrict__ Wg,
                      const float* __restrict__ bg,
                      float* __restrict__ out)
{
    __shared__ __align__(16) unsigned char pool[POOLSZ];
    __bf16 (*ldsT)[TPAD]   = (__bf16 (*)[TPAD])(pool + OFF_T);     // [256][TPAD]
    __bf16 (*ldsK)[Dd]     = (__bf16 (*)[Dd])  (pool + OFF_ROW);   // [32][256]
    __bf16 (*ldsQ)[Dd]     = (__bf16 (*)[Dd])  (pool + OFF_Q);     // [32][256]
    __bf16 (*ldsO)[Dd]     = (__bf16 (*)[Dd])  (pool + OFF_ROW);   // alias of ldsK (epilogue)
    __bf16 (*ldsP)[16][32] = (__bf16 (*)[16][32])(pool + OFF_P);   // [2][16][32]
    float  (*ldsGP)[ROWS]  = (float (*)[ROWS]) (pool + OFF_GP);    // [2][32]
    float  *ldsG           = (float*)          (pool + OFF_G);     // [32]

    const int tid  = threadIdx.x;
    const int wave = tid >> 5;
    const int lane = tid & 31;
    const int hi   = lane >> 4;     // which 16-lane half
    const int ln   = lane & 15;

    const int batch = blockIdx.x / (Ld / ROWS);
    const int row0  = (blockIdx.x % (Ld / ROWS)) * ROWS;  // query row base in L
    const int wrow0 = wave * 16;                          // wave row base in WG

    const float* c1b = c1 + (size_t)batch * Ld * Dd;
    const float* c2b = c2 + (size_t)batch * Ld * Dd;
    const unsigned char* cmb = cmask + (size_t)batch * Ld;

    const float NEG = -__builtin_inff();

    // ---- stage query tile (c2) into LDS as bf16 (row-major only) ----
    for (int idx = tid; idx < ROWS * Dd / 4; idx += TPB) {
        int r = idx / (Dd / 4), c4 = (idx % (Dd / 4)) * 4;
        const float4 v = *(const float4*)(c2b + (size_t)(row0 + r) * Dd + c4);
        v4bf pk = { (__bf16)v.x, (__bf16)v.y, (__bf16)v.z, (__bf16)v.w };
        *(v4bf*)&ldsQ[r][c4] = pk;
    }

    // O accumulators: 16 d-tiles of 16x16 f32 (= 16 rows x 256 cols)
    v8f o[16];
    #pragma unroll
    for (int n = 0; n < 16; ++n) o[n] = (v8f)0.f;
    float rmax[8], lsum[8];
    #pragma unroll
    for (int v = 0; v < 8; ++v) { rmax[v] = NEG; lsum[v] = 0.f; }

    // =================== flash pass over key chunks ===================
    for (int jc = 0; jc < Ld / NCH; ++jc) {
        __syncthreads();
        // stage c1 chunk: row-major (vector stores) + transposed copy
        for (int idx = tid; idx < NCH * Dd / 4; idx += TPB) {
            int r = idx / (Dd / 4), c4 = (idx % (Dd / 4)) * 4;
            const float4 v = *(const float4*)(c1b + (size_t)(jc * NCH + r) * Dd + c4);
            __bf16 b0 = (__bf16)v.x, b1 = (__bf16)v.y, b2 = (__bf16)v.z, b3 = (__bf16)v.w;
            v4bf pk = { b0, b1, b2, b3 };
            *(v4bf*)&ldsK[r][c4] = pk;
            ldsT[c4 + 0][r] = b0; ldsT[c4 + 1][r] = b1;
            ldsT[c4 + 2][r] = b2; ldsT[c4 + 3][r] = b3;
        }
        __syncthreads();

        // prefetch next chunk (overlaps with WMMA work below)
        {
            const int jn = (jc + 1) & (Ld / NCH - 1);
            const char* nb = (const char*)(c1b + (size_t)jn * NCH * Dd);
            #pragma unroll
            for (int pf = 0; pf < 4; ++pf)
                __builtin_prefetch(nb + tid * 128 + pf * 8192, 0, 1);
        }

        // ---- S(16x32) = q_tile(16xD) @ c1_chunk^T(Dx32) ----
        v8f s0 = (v8f)0.f, s1 = (v8f)0.f;
        #pragma unroll
        for (int ks = 0; ks < Dd / 32; ++ks) {
            v16bf a;
            {   // A frag: M=ln, K per ISA 16-bit A layout (two contiguous 16B runs)
                const v8bf alo = *(const v8bf*)&ldsQ[wrow0 + ln][ks * 32 + hi * 8];
                const v8bf ahi = *(const v8bf*)&ldsQ[wrow0 + ln][ks * 32 + 16 + hi * 8];
                #pragma unroll
                for (int i = 0; i < 8; ++i) { a[i] = alo[i]; a[8 + i] = ahi[i]; }
            }
            v16bf b0, b1;
            {   // B frag: N=ln (key), K per ISA 16-bit layout (two 16B runs, same as A)
                const v8bf l0 = *(const v8bf*)&ldsK[ln     ][ks * 32 + hi * 8];
                const v8bf h0 = *(const v8bf*)&ldsK[ln     ][ks * 32 + 16 + hi * 8];
                const v8bf l1 = *(const v8bf*)&ldsK[16 + ln][ks * 32 + hi * 8];
                const v8bf h1 = *(const v8bf*)&ldsK[16 + ln][ks * 32 + 16 + hi * 8];
                #pragma unroll
                for (int i = 0; i < 8; ++i) {
                    b0[i] = l0[i]; b0[8 + i] = h0[i];
                    b1[i] = l1[i]; b1[8 + i] = h1[i];
                }
            }
            s0 = __builtin_amdgcn_wmma_f32_16x16x32_bf16(false, a, false, b0, (short)0, s0, false, false);
            s1 = __builtin_amdgcn_wmma_f32_16x16x32_bf16(false, a, false, b1, (short)0, s1, false, false);
        }

        // ---- masks + online softmax (row m = v + 8*hi lives in one 16-lane half) ----
        const bool mk0 = cmb[jc * NCH + ln] != 0;
        const bool mk1 = cmb[jc * NCH + 16 + ln] != 0;
        const int  n0g = jc * NCH + ln;
        const int  n1g = jc * NCH + 16 + ln;
        float cmax[8];
        #pragma unroll
        for (int v = 0; v < 8; ++v) {
            const int qg = row0 + wrow0 + v + 8 * hi;   // global query row (diag mask)
            float x0 = (mk0 || (n0g == qg)) ? NEG : s0[v];
            float x1 = (mk1 || (n1g == qg)) ? NEG : s1[v];
            s0[v] = x0; s1[v] = x1;
            float cm = fmaxf(x0, x1);
            #pragma unroll
            for (int off = 1; off < 16; off <<= 1)
                cm = fmaxf(cm, __shfl_xor(cm, off, 32));
            cmax[v] = cm;
        }
        float f[8];
        #pragma unroll
        for (int v = 0; v < 8; ++v) {
            float nm    = fmaxf(rmax[v], cmax[v]);
            float shift = (nm == NEG) ? 0.f : nm;
            f[v]    = __expf(rmax[v] - shift);
            rmax[v] = nm;
            float p0 = __expf(s0[v] - shift);
            float p1 = __expf(s1[v] - shift);
            float ps = p0 + p1;
            #pragma unroll
            for (int off = 1; off < 16; off <<= 1)
                ps += __shfl_xor(ps, off, 32);
            lsum[v] = lsum[v] * f[v] + ps;
            ldsP[wave][v + 8 * hi][ln]      = (__bf16)p0;   // re-layout P via LDS
            ldsP[wave][v + 8 * hi][16 + ln] = (__bf16)p1;
        }
        #pragma unroll
        for (int n = 0; n < 16; ++n)
            #pragma unroll
            for (int v = 0; v < 8; ++v)
                o[n][v] *= f[v];
        __syncthreads();   // P visible cross-lane; waves stay in lockstep

        // ---- O += P(16x32) @ c1_chunk(32xD), B frags from transposed copy ----
        v16bf pa;
        {
            const v8bf plo = *(const v8bf*)&ldsP[wave][ln][hi * 8];
            const v8bf phi = *(const v8bf*)&ldsP[wave][ln][16 + hi * 8];
            #pragma unroll
            for (int i = 0; i < 8; ++i) { pa[i] = plo[i]; pa[8 + i] = phi[i]; }
        }
        #pragma unroll
        for (int n = 0; n < 16; ++n) {
            v16bf b;   // B[k][ncol] = c1T[ncol][k], K per ISA layout (same as A)
            const v8bf l0 = *(const v8bf*)&ldsT[n * 16 + ln][hi * 8];
            const v8bf h0 = *(const v8bf*)&ldsT[n * 16 + ln][16 + hi * 8];
            #pragma unroll
            for (int i = 0; i < 8; ++i) { b[i] = l0[i]; b[8 + i] = h0[i]; }
            o[n] = __builtin_amdgcn_wmma_f32_16x16x32_bf16(false, pa, false, b, (short)0, o[n], false, false);
        }
    }

    // ---- normalize and park aug_c2 in LDS (aliases chunk row buffer) ----
    __syncthreads();
    float inv[8];
    #pragma unroll
    for (int v = 0; v < 8; ++v) inv[v] = 1.f / lsum[v];
    #pragma unroll
    for (int n = 0; n < 16; ++n)
        #pragma unroll
        for (int v = 0; v < 8; ++v)
            ldsO[wrow0 + v + 8 * hi][n * 16 + ln] = (__bf16)(o[n][v] * inv[v]);
    __syncthreads();

    // ---- gate: g = sigmoid(z . W_g + b_g), 2 threads per row ----
    {
        const int r = tid & (ROWS - 1);
        const int part = tid >> 5;
        float acc = 0.f;
        for (int k = part * (KF / 2); k < (part + 1) * (KF / 2); ++k) {
            int q = k >> 8, i = k & 255;
            float x = (float)ldsQ[r][i];
            float a = (float)ldsO[r][i];
            float zv = (q == 0) ? x : (q == 1) ? a : (q == 2) ? x * a : x - a;
            acc += zv * Wg[k];
        }
        ldsGP[part][r] = acc;
    }
    __syncthreads();
    if (tid < ROWS) {
        float s = ldsGP[0][tid] + ldsGP[1][tid] + bg[0];
        ldsG[tid] = 1.f / (1.f + __expf(-s));
    }

    // ---- fusion GEMM: fc(16x256) = z(16x1024) @ W_f(1024x256), WMMA ----
    v8f fc[16];
    #pragma unroll
    for (int n = 0; n < 16; ++n) fc[n] = (v8f)0.f;

    for (int kk = 0; kk < KF / 32; ++kk) {
        __syncthreads();
        // stage W_f chunk transposed only: WfT[col][k]
        for (int idx = tid; idx < 32 * Dd / 4; idx += TPB) {
            int r = idx / (Dd / 4), c4 = (idx % (Dd / 4)) * 4;
            const float4 v = *(const float4*)(Wf + (size_t)(kk * 32 + r) * Dd + c4);
            ldsT[c4 + 0][r] = (__bf16)v.x; ldsT[c4 + 1][r] = (__bf16)v.y;
            ldsT[c4 + 2][r] = (__bf16)v.z; ldsT[c4 + 3][r] = (__bf16)v.w;
        }
        __syncthreads();

        // prefetch next W_f chunk
        {
            const int kn = (kk + 1) & (KF / 32 - 1);
            const char* nb = (const char*)(Wf + (size_t)kn * 32 * Dd);
            #pragma unroll
            for (int pf = 0; pf < 4; ++pf)
                __builtin_prefetch(nb + tid * 128 + pf * 8192, 0, 1);
        }

        v16bf za;                                  // build z A-fragment on the fly
        const int zr = wrow0 + ln;
        #pragma unroll
        for (int h2 = 0; h2 < 16; ++h2) {
            int r5 = h2 >> 1, p = h2 & 1;
            int kl = ((r5 & 3) * 2 + p) + ((r5 >> 2) * 16) + hi * 8;   // ISA A layout
            int kg = kk * 32 + kl;
            int q = kg >> 8, i = kg & 255;
            float x = (float)ldsQ[zr][i];
            float a = (float)ldsO[zr][i];
            float zv = (q == 0) ? x : (q == 1) ? a : (q == 2) ? x * a : x - a;
            za[h2] = (__bf16)zv;
        }
        #pragma unroll
        for (int n = 0; n < 16; ++n) {
            v16bf b;   // B[k][col] = WfT[col][k], K per ISA layout (same as A)
            const v8bf l0 = *(const v8bf*)&ldsT[n * 16 + ln][hi * 8];
            const v8bf h0 = *(const v8bf*)&ldsT[n * 16 + ln][16 + hi * 8];
            #pragma unroll
            for (int i = 0; i < 8; ++i) { b[i] = l0[i]; b[8 + i] = h0[i]; }
            fc[n] = __builtin_amdgcn_wmma_f32_16x16x32_bf16(false, za, false, b, (short)0, fc[n], false, false);
        }
    }
    __syncthreads();

    // ---- epilogue: out = g*tanh(fc + b_f) + (1-g)*x ----
    #pragma unroll
    for (int n = 0; n < 16; ++n) {
        #pragma unroll
        for (int v = 0; v < 8; ++v) {
            const int wr  = wrow0 + v + 8 * hi;
            const int col = n * 16 + ln;
            const float g = ldsG[wr];
            const float t = tanhf(fc[n][v] + bfv[col]);
            const float x = c2b[(size_t)(row0 + wr) * Dd + col];
            out[(size_t)batch * Ld * Dd + (size_t)(row0 + wr) * Dd + col] =
                g * t + (1.f - g) * x;
        }
    }
}

extern "C" void kernel_launch(void* const* d_in, const int* in_sizes, int n_in,
                              void* d_out, int out_size, void* d_ws, size_t ws_size,
                              hipStream_t stream) {
    (void)in_sizes; (void)n_in; (void)out_size; (void)d_ws; (void)ws_size;
    const float* c1 = (const float*)d_in[0];
    const float* c2 = (const float*)d_in[1];
    const unsigned char* cm = (const unsigned char*)d_in[2];   // bool mask, 1 byte/elem
    const float* Wf  = (const float*)d_in[3];
    const float* bfv = (const float*)d_in[4];
    const float* Wg  = (const float*)d_in[5];
    const float* bg  = (const float*)d_in[6];
    float* out = (float*)d_out;

    dim3 grid(Bn * (Ld / ROWS));   // 512 workgroups
    dim3 block(TPB);               // 64 threads = 2 wave32
    hipLaunchKernelGGL(aoa_fused_kernel, grid, block, 0, stream,
                       c1, c2, cm, Wf, bfv, Wg, bg, out);
}
